// SpectralConvS2_27530740367617
// MI455X (gfx1250) — matrix operand order repaired and checked by reading, not compiled, not run on hardware
//
#include <hip/hip_runtime.h>

typedef __attribute__((ext_vector_type(16))) _Float16 v16h;
typedef __attribute__((ext_vector_type(8)))  float    v8f;

struct __align__(16) U16x4 { unsigned u[4]; };

constexpr int BM = 128, BN = 64, BK = 32, LDSP = 40;   // LDS pitch in halfs: 80B, 16B-aligned

// ---------------------------------------------------------------------------
// DFT factor matrix generator:
//   re = s*cos(2pi*((r*c)%modN)/modN),  im = imSign*s*sin(...),  s = norm*(harm&&r>0 ? 2:1)
// ---------------------------------------------------------------------------
__global__ void gen_dft(_Float16* __restrict__ re, _Float16* __restrict__ im,
                        int rows, int cols, int modN, float norm, float imSign, int harm) {
    int idx = blockIdx.x * blockDim.x + threadIdx.x;
    if (idx >= rows * cols) return;
    int r = idx / cols, c = idx % cols;
    int t = (int)(((long long)r * (long long)c) % modN);
    float ph = 6.28318530717958647692f * (float)t / (float)modN;
    float s = norm;
    if (harm && r > 0) s *= 2.0f;
    re[idx] = (_Float16)(s * cosf(ph));
    im[idx] = (_Float16)(imSign * s * sinf(ph));
}

__global__ void conv_f16(const float* __restrict__ src, _Float16* __restrict__ dst, long long n) {
    long long i = (long long)blockIdx.x * blockDim.x + threadIdx.x;
    if (i < n) dst[i] = (_Float16)src[i];
}

// weight [i][o][l][2] fp32 -> Wre/Wim [l][i][o] f16, scaled by 256 (undone in inv col DFT)
__global__ void pack_w(const float* __restrict__ w, _Float16* __restrict__ wre,
                       _Float16* __restrict__ wim) {
    int idx = blockIdx.x * blockDim.x + threadIdx.x;
    if (idx >= 64 * 64 * 240) return;
    int l  = idx % 240;
    int io = idx / 240;               // io = i*64 + o
    long long src = ((long long)io * 240 + l) * 2;
    int i = io >> 6, o = io & 63;
    long long dst = ((long long)l * 64 + i) * 64 + o;
    wre[dst] = (_Float16)(256.0f * w[src]);
    wim[dst] = (_Float16)(256.0f * w[src + 1]);
}

// X2 [b*64+i][l][m] -> X2p [l][b*241+m][i]   total 240*964*64 per plane
__global__ void pack_p1(const _Float16* __restrict__ x2re, const _Float16* __restrict__ x2im,
                        _Float16* __restrict__ pre, _Float16* __restrict__ pim) {
    long long idx = (long long)blockIdx.x * blockDim.x + threadIdx.x;
    if (idx >= (long long)240 * 964 * 64) return;
    int i = (int)(idx & 63);
    long long t = idx >> 6;
    int r = (int)(t % 964);
    int l = (int)(t / 964);
    int m = r % 241, b = r / 241;
    long long src = ((long long)(b * 64 + i) * 240 + l) * 241 + m;
    pre[idx] = x2re[src];
    pim[idx] = x2im[src];
}

// Z [lat][(b*241+m)*64+o] -> Zt [(b*64+o)*360+lat][m], m padded 241->256 with zeros
__global__ void pack_p2(const _Float16* __restrict__ zre, const _Float16* __restrict__ zim,
                        _Float16* __restrict__ tre, _Float16* __restrict__ tim) {
    long long idx = (long long)blockIdx.x * blockDim.x + threadIdx.x;
    if (idx >= (long long)92160 * 256) return;
    int m = (int)(idx & 255);
    long long row = idx >> 8;
    int lat = (int)(row % 360);
    int bo  = (int)(row / 360);
    int b = bo >> 6, o = bo & 63;
    _Float16 vr = (_Float16)0.0f, vi = (_Float16)0.0f;
    if (m < 241) {
        long long src = (long long)lat * 61696 + ((long long)b * 241 + m) * 64 + o;
        vr = zre[src]; vi = zim[src];
    }
    tre[idx] = vr;
    tim[idx] = vi;
}

// ---------------------------------------------------------------------------
// WMMA f16 fragment helpers (CDNA5 ISA 7.12.2, wave32)
// A 16x32: lane<16 -> K 0..7 / 16..23 ; lane>=16 -> K 8..15 / 24..31 (row = lane&15)
// B 32x16 stored transposed in LDS [n][k]: col = lane&15, lane<16 -> K 0..7 / 16..23 ; lane>=16 -> K 8..15 / 24..31
// ---------------------------------------------------------------------------
__device__ inline v16h load_fragA(const _Float16* As, int row, int hl) {
    union { v16h v; U16x4 q[2]; } f;
    f.q[0] = *(const U16x4*)&As[row * LDSP + hl * 8];
    f.q[1] = *(const U16x4*)&As[row * LDSP + 16 + hl * 8];
    return f.v;
}
__device__ inline v16h load_fragB(const _Float16* Bs, int col, int hl) {
    union { v16h v; U16x4 q[2]; } f;
    f.q[0] = *(const U16x4*)&Bs[col * LDSP + hl * 8];
    f.q[1] = *(const U16x4*)&Bs[col * LDSP + 16 + hl * 8];
    return f.v;
}
__device__ inline v16h negh(v16h a) {   // f16 sign flip (NEG modifier illegal on f16 A/B)
    union { v16h v; unsigned u[8]; } f; f.v = a;
#pragma unroll
    for (int j = 0; j < 8; ++j) f.u[j] ^= 0x80008000u;
    return f.v;
}

// ---------------------------------------------------------------------------
// Batched planar-complex GEMM: Cre = Are*Bre - Aim*Bim ; Cim = Are*Bim + Aim*Bre
// A row-major lda=K (K % 8 == 0 required), B row-major ldb=N, C row-major ldc=N.
// Double-buffered LDS; A tiles via global_load_async_to_lds_b128 (ASYNCcnt),
// B tiles via VGPR transpose + ds stores; fp32 WMMA accumulate.
// ---------------------------------------------------------------------------
template<bool HAS_AIM, bool HAS_CIM, bool OUT_F32>
__global__ __launch_bounds__(256)
void cgemm_wmma(const _Float16* __restrict__ AreG, const _Float16* __restrict__ AimG,
                const _Float16* __restrict__ BreG, const _Float16* __restrict__ BimG,
                void* __restrict__ CrePtr, void* __restrict__ CimPtr,
                int M, int N, int K, long long sA, long long sB, long long sC) {
    __shared__ __align__(16) _Float16 AsRe[2 * BM * LDSP];
    __shared__ __align__(16) _Float16 AsIm[HAS_AIM ? 2 * BM * LDSP : 16];
    __shared__ __align__(16) _Float16 BsRe[2 * BN * LDSP];
    __shared__ __align__(16) _Float16 BsIm[2 * BN * LDSP];

    const int z   = blockIdx.z;
    const int m0  = blockIdx.y * BM;
    const int n0  = blockIdx.x * BN;
    const int tid = threadIdx.x;
    const int lane = tid & 31;
    const int wid  = tid >> 5;
    const int wm = wid >> 1;       // 0..3 -> 32-row strip
    const int wn = wid & 1;        // 0..1 -> 32-col strip
    const int hl  = lane >> 4;
    const int l16 = lane & 15;

    const _Float16* Are = AreG + (size_t)z * (size_t)sA;
    const _Float16* Aim = AimG ? AimG + (size_t)z * (size_t)sA : nullptr;
    const _Float16* Bre = BreG + (size_t)z * (size_t)sB;
    const _Float16* Bim = BimG + (size_t)z * (size_t)sB;

    const v8f z8 = {0.f, 0.f, 0.f, 0.f, 0.f, 0.f, 0.f, 0.f};
    v8f cre[2][2], cim[2][2];
#pragma unroll
    for (int a = 0; a < 2; ++a)
#pragma unroll
        for (int b = 0; b < 2; ++b) { cre[a][b] = z8; if constexpr (HAS_CIM) cim[a][b] = z8; }

    // ---- tile loader: A async-to-LDS, B sync with transpose; buffer b in {0,1} ----
    auto load_tiles = [&](int kt, int b) {
        const int k0 = kt * BK;
        _Float16* AsR = &AsRe[b * BM * LDSP];
        _Float16* AsI = &AsIm[HAS_AIM ? b * BM * LDSP : 0];
        _Float16* BsR = &BsRe[b * BN * LDSP];
        _Float16* BsI = &BsIm[b * BN * LDSP];
        // A tiles: 128x32 halfs = 512 16B-chunks, 2 per thread
#pragma unroll
        for (int t2 = 0; t2 < 2; ++t2) {
            int chunk = tid + t2 * 256;
            int r  = chunk >> 2;
            int c8 = (chunk & 3) << 3;
            int gr = m0 + r;
            int gk = k0 + c8;
            _Float16* dR = &AsR[r * LDSP + c8];
            if (gr < M && gk < K) {        // K%8==0 -> full 16B chunk in-bounds
                unsigned lre = (unsigned)(size_t)dR;           // flat ptr low bits = LDS offset
                unsigned long long gre = (unsigned long long)(Are + (size_t)gr * K + gk);
                asm volatile("global_load_async_to_lds_b128 %0, %1, off"
                             :: "v"(lre), "v"(gre) : "memory");
                if constexpr (HAS_AIM) {
                    unsigned lim = (unsigned)(size_t)&AsI[r * LDSP + c8];
                    unsigned long long gim = (unsigned long long)(Aim + (size_t)gr * K + gk);
                    asm volatile("global_load_async_to_lds_b128 %0, %1, off"
                                 :: "v"(lim), "v"(gim) : "memory");
                }
            } else {                        // out-of-range rows / K tail: zero-fill
#pragma unroll
                for (int j = 0; j < 8; ++j) {
                    bool in = (gr < M) && (gk + j < K);
                    dR[j] = in ? Are[(size_t)gr * K + gk + j] : (_Float16)0.0f;
                    if constexpr (HAS_AIM)
                        AsI[r * LDSP + c8 + j] =
                            in ? Aim[(size_t)gr * K + gk + j] : (_Float16)0.0f;
                }
            }
        }
        // B tile: 32x64, transposed into LDS [n][k]; 1 8-half chunk per thread
        {
            int kk = tid >> 3;
            int c8 = (tid & 7) << 3;
            int gk = k0 + kk;
            int gn = n0 + c8;
            bool ok = gk < K;
            _Float16 vr[8], vi[8];
            if (((N & 7) == 0) && ok && gn < N) {
                union { U16x4 q; _Float16 h[8]; } ur, ui;
                ur.q = *(const U16x4*)&Bre[(size_t)gk * N + gn];
                ui.q = *(const U16x4*)&Bim[(size_t)gk * N + gn];
#pragma unroll
                for (int j = 0; j < 8; ++j) { vr[j] = ur.h[j]; vi[j] = ui.h[j]; }
                if (gk + BK < K)            // prefetch tile kt+1 (pipelined -> t+2 overall)
                    __builtin_prefetch(&Bre[(size_t)(gk + BK) * N + gn], 0, 0);
            } else {
#pragma unroll
                for (int j = 0; j < 8; ++j) {
                    bool in = ok && (gn + j < N);
                    vr[j] = in ? Bre[(size_t)gk * N + gn + j] : (_Float16)0.0f;
                    vi[j] = in ? Bim[(size_t)gk * N + gn + j] : (_Float16)0.0f;
                }
            }
#pragma unroll
            for (int j = 0; j < 8; ++j) {
                BsR[(c8 + j) * LDSP + kk] = vr[j];
                BsI[(c8 + j) * LDSP + kk] = vi[j];
            }
        }
    };

    // ---- compute one k-tile from buffer b: each wave = 32x32 = 2x2 WMMA tiles ----
    auto compute = [&](int b) {
        const _Float16* AsR = &AsRe[b * BM * LDSP];
        const _Float16* AsI = &AsIm[HAS_AIM ? b * BM * LDSP : 0];
        const _Float16* BsR = &BsRe[b * BN * LDSP];
        const _Float16* BsI = &BsIm[b * BN * LDSP];
        v16h are[2], aim[2], ain[2];
#pragma unroll
        for (int mt = 0; mt < 2; ++mt) {
            int row = wm * 32 + mt * 16 + l16;
            are[mt] = load_fragA(AsR, row, hl);
            if constexpr (HAS_AIM) {
                aim[mt] = load_fragA(AsI, row, hl);
                ain[mt] = negh(aim[mt]);
            }
        }
#pragma unroll
        for (int nt = 0; nt < 2; ++nt) {
            int col = wn * 32 + nt * 16 + l16;
            v16h bre = load_fragB(BsR, col, hl);
            v16h bim = load_fragB(BsI, col, hl);
#pragma unroll
            for (int mt = 0; mt < 2; ++mt) {
                cre[mt][nt] = __builtin_amdgcn_wmma_f32_16x16x32_f16(
                    false, are[mt], false, bre, (short)0, cre[mt][nt], false, false);
                if constexpr (HAS_AIM)
                    cre[mt][nt] = __builtin_amdgcn_wmma_f32_16x16x32_f16(
                        false, ain[mt], false, bim, (short)0, cre[mt][nt], false, false);
                if constexpr (HAS_CIM) {
                    cim[mt][nt] = __builtin_amdgcn_wmma_f32_16x16x32_f16(
                        false, are[mt], false, bim, (short)0, cim[mt][nt], false, false);
                    if constexpr (HAS_AIM)
                        cim[mt][nt] = __builtin_amdgcn_wmma_f32_16x16x32_f16(
                            false, aim[mt], false, bre, (short)0, cim[mt][nt], false, false);
                }
            }
        }
    };

    // ---- software pipeline: async loads for tile t+1 overlap WMMA on tile t ----
    const int nk = (K + BK - 1) / BK;
    load_tiles(0, 0);
    int b = 0;
    for (int t = 0; t < nk; ++t) {
        asm volatile("s_wait_asynccnt 0x0" ::: "memory");  // this wave's async A-tile done
        __syncthreads();                                    // all waves' tile-t data visible
        if (t + 1 < nk) load_tiles(t + 1, b ^ 1);
        compute(b);
        b ^= 1;
    }

    // ---- store (C layout: vgpr r -> row = r + 8*hl, col = lane&15) ----
#pragma unroll
    for (int mt = 0; mt < 2; ++mt)
#pragma unroll
        for (int nt = 0; nt < 2; ++nt) {
            int row0 = m0 + wm * 32 + mt * 16 + hl * 8;
            int col  = n0 + wn * 32 + nt * 16 + l16;
            if (col < N) {
#pragma unroll
                for (int r = 0; r < 8; ++r) {
                    int row = row0 + r;
                    if (row < M) {
                        size_t off = (size_t)z * (size_t)sC + (size_t)row * N + col;
                        if constexpr (OUT_F32) ((float*)CrePtr)[off] = cre[mt][nt][r];
                        else                   ((_Float16*)CrePtr)[off] = (_Float16)cre[mt][nt][r];
                        if constexpr (HAS_CIM) {
                            if constexpr (OUT_F32) ((float*)CimPtr)[off] = cim[mt][nt][r];
                            else                   ((_Float16*)CimPtr)[off] = (_Float16)cim[mt][nt][r];
                        }
                    }
                }
            }
        }
}

// ---------------------------------------------------------------------------
extern "C" void kernel_launch(void* const* d_in, const int* in_sizes, int n_in,
                              void* d_out, int out_size, void* d_ws, size_t ws_size,
                              hipStream_t stream) {
    (void)in_sizes; (void)n_in; (void)out_size; (void)ws_size;
    const float* x = (const float*)d_in[0];   // [4,64,360,720]
    const float* w = (const float*)d_in[1];   // [64,64,240,2]
    float* outp = (float*)d_out;              // [4,64,360,720] out + residual copy of x

    // ---- workspace arena (counts in f16 halves) ----
    size_t off = 0;
    auto take = [&](size_t n) { size_t c = off; off += (n + 127) & ~(size_t)127; return c; };
    const size_t oRre = take(720 * 241),  oRim = take(720 * 241);     // row fwd DFT
    const size_t oCre = take(240 * 360),  oCim = take(240 * 360);     // col fwd DFT
    const size_t oAre = take(360 * 240),  oAim = take(360 * 240);     // col inv DFT
    const size_t oSre = take(256 * 720),  oSim = take(256 * 720);     // row inv DFT (K padded)
    const size_t oWre = take(240 * 64 * 64), oWim = take(240 * 64 * 64);
    const size_t oX1re = take((size_t)92160 * 241), oX1im = take((size_t)92160 * 241);
    const size_t oX2re = take((size_t)256 * 240 * 241), oX2im = take((size_t)256 * 240 * 241);
    const size_t oPre  = take((size_t)240 * 964 * 64),  oPim  = take((size_t)240 * 964 * 64);
    const size_t oX3re = take((size_t)240 * 964 * 64),  oX3im = take((size_t)240 * 964 * 64);
    // overlays of dead regions: xh spans X2+P+X3 (live only during stage A);
    // Z overlays X1; Zt overlays X2(+X2im) and P(+Pim) regions.
    const size_t oXh  = oX2re;                 // 92160*720 = 66.4M halfs <= 88.8M capacity
    const size_t oZre = oX1re, oZim = oX1im;
    const size_t oZtre = oX2re, oZtim = oPre;  // 92160*256 = 23.6M halfs each
    auto H = [&](size_t o) { return (_Float16*)d_ws + o; };

    const float inv720 = 0.03726779962499649f;
    const float inv360 = 0.05270462766947299f;

    // ---- residual: second half of d_out = x ----
    hipMemcpyAsync(outp + (size_t)4 * 64 * 360 * 720, x,
                   (size_t)4 * 64 * 360 * 720 * sizeof(float),
                   hipMemcpyDeviceToDevice, stream);

    // ---- DFT factor matrices, weight pack, x -> f16 ----
    auto g1 = [](long long n) { return dim3((unsigned)((n + 255) / 256)); };
    gen_dft<<<g1(720 * 241), 256, 0, stream>>>(H(oRre), H(oRim), 720, 241, 720, inv720, -1.0f, 0);
    gen_dft<<<g1(240 * 360), 256, 0, stream>>>(H(oCre), H(oCim), 240, 360, 360, inv360, -1.0f, 0);
    gen_dft<<<g1(360 * 240), 256, 0, stream>>>(H(oAre), H(oAim), 360, 240, 360,
                                               inv360 / 256.0f, +1.0f, 0);  // undo W scale
    gen_dft<<<g1(256 * 720), 256, 0, stream>>>(H(oSre), H(oSim), 256, 720, 720, inv720, +1.0f, 1);
    pack_w<<<g1(64 * 64 * 240), 256, 0, stream>>>(w, H(oWre), H(oWim));
    conv_f16<<<g1((long long)92160 * 720), 256, 0, stream>>>(x, H(oXh), (long long)92160 * 720);

    // ---- stage A: row rfft.  X1 = xh * R ----
    cgemm_wmma<false, true, false><<<dim3(4, 720, 1), 256, 0, stream>>>(
        H(oXh), nullptr, H(oRre), H(oRim), H(oX1re), H(oX1im),
        92160, 241, 720, 0, 0, 0);

    // ---- stage B: col DFT (batched over 256 images).  X2 = Ccol * X1_img ----
    cgemm_wmma<true, true, false><<<dim3(4, 2, 256), 256, 0, stream>>>(
        H(oCre), H(oCim), H(oX1re), H(oX1im), H(oX2re), H(oX2im),
        240, 241, 360, 0, (long long)360 * 241, (long long)240 * 241);

    // ---- pack: X2 [b,i][l,m] -> X2p [l][(b,m)][i] ----
    pack_p1<<<g1((long long)240 * 964 * 64), 256, 0, stream>>>(H(oX2re), H(oX2im),
                                                               H(oPre), H(oPim));

    // ---- stage C: per-l complex channel mix (batched over 240 l).  X3 = X2p * W_l ----
    cgemm_wmma<true, true, false><<<dim3(1, 8, 240), 256, 0, stream>>>(
        H(oPre), H(oPim), H(oWre), H(oWim), H(oX3re), H(oX3im),
        964, 64, 64, (long long)964 * 64, 4096, (long long)964 * 64);

    // ---- stage D: col inverse DFT, one big GEMM.  Z = Acol * X3 ----
    cgemm_wmma<true, true, false><<<dim3(964, 3, 1), 256, 0, stream>>>(
        H(oAre), H(oAim), H(oX3re), H(oX3im), H(oZre), H(oZim),
        360, 61696, 240, 0, 0, 0);

    // ---- pack: Z [lat][(b,m,o)] -> Zt [(b,o,lat)][m] (m padded to 256) ----
    pack_p2<<<g1((long long)92160 * 256), 256, 0, stream>>>(H(oZre), H(oZim),
                                                            H(oZtre), H(oZtim));

    // ---- stage E: row irfft, real fp32 output.  out = Zre*Sre - Zim*Sim ----
    cgemm_wmma<true, false, true><<<dim3(12, 720, 1), 256, 0, stream>>>(
        H(oZtre), H(oZtim), H(oSre), H(oSim), outp, nullptr,
        92160, 720, 256, 0, 0, 0);
}
